// SNN_59588376265070
// MI455X (gfx1250) — compile-verified, matrix operand order checked by reading, not run on hardware
//
#include <hip/hip_runtime.h>

// ---------------------------------------------------------------------------
// Problem constants (from reference): N=256, T=100, C=700, H=1024, NC=20
// ---------------------------------------------------------------------------
#define Tn 100
#define Nn 256
#define Cc 700
#define Cp 704          // C padded to multiple of 32 (WMMA K-chunk)
#define Hh 1024
#define NCLS 20

typedef __attribute__((ext_vector_type(16))) __bf16        v16bf;
typedef __attribute__((ext_vector_type(8)))  float         v8f;

union ABFrag { v16bf v; unsigned int u[8]; };

// ---------------------------------------------------------------------------
// CDNA5 async global->LDS (ISA 15.18.3 op 98) + ASYNCcnt wait (ISA 9.6 op 74)
// ---------------------------------------------------------------------------
__device__ __forceinline__ void async_load_b128(unsigned ldsAddr, const __bf16* gaddr) {
  asm volatile("global_load_async_to_lds_b128 %0, %1, off"
               :: "v"(ldsAddr), "v"(gaddr) : "memory");
}
__device__ __forceinline__ void wait_async0() {
  asm volatile("s_wait_asynccnt 0x0" ::: "memory");
}
// low 32 bits of a flat pointer to __shared__ == workgroup-relative LDS offset
__device__ __forceinline__ unsigned lds_off(const void* p) {
  return (unsigned)(unsigned long long)p;
}

// ---------------------------------------------------------------------------
// Conversion / init kernels
// ---------------------------------------------------------------------------
// W [K,N] fp32 -> Wt [N,Kp] bf16 (transposed + zero-padded K)
__global__ void k_transpose_w(const float* __restrict__ src, __bf16* __restrict__ dst,
                              int K, int N, int Kp) {
  int i = blockIdx.x * 256 + threadIdx.x;
  if (i >= N * Kp) return;
  int n = i / Kp, k = i - n * Kp;
  dst[i] = (k < K) ? (__bf16)src[(long)k * N + n] : (__bf16)0.0f;
}

// x [N,T,700] fp32 -> xbf [T,N,704] bf16 (t-major + zero pad)
__global__ void k_conv_x(const float* __restrict__ x, __bf16* __restrict__ dst) {
  long i = (long)blockIdx.x * 256 + threadIdx.x;
  if (i >= (long)Tn * Nn * Cp) return;
  int  c = (int)(i % Cp);
  long r = i / Cp;
  int  n = (int)(r % Nn);
  int  t = (int)(r / Nn);
  dst[i] = (c < Cc) ? (__bf16)x[((long)n * Tn + t) * Cc + c] : (__bf16)0.0f;
}

__global__ void k_zero_f32(float* p, int n) {
  int i = blockIdx.x * 256 + threadIdx.x;
  if (i < n) p[i] = 0.0f;
}
__global__ void k_zero_bf16(__bf16* p, int n) {
  int i = blockIdx.x * 256 + threadIdx.x;
  if (i < n) p[i] = (__bf16)0.0f;
}

// ---------------------------------------------------------------------------
// Fused WMMA GEMM: C[M,ldc] = A[M,K](bf16) x Bt[ldc,K]^T(bf16) + bias
//   Bt is the PRE-TRANSPOSED weight: Bt[n][k]  (A and B tiles stage alike)
//   mode 0: outF[row,col] = acc + bias[col]
//   mode 1: LIF epilogue (TAU=2, VTH=1, hard reset):
//             pre = acc + bias + (addend? addend[row,col]:0)
//             h = v + (pre-v)*0.5 ; s = (h-1>=0) ; v = h*(1-s) ; outS = bf16(s)
// Block: 256 thr = 8 waves (2M x 4N). Wave tile: 32 x (NI*16) -> 2xNI WMMAs.
//   NI=2: block tile 64x128 (recurrent steps: keeps 32 WGs of parallelism)
//   NI=4: block tile 64x256 (big batched GEMMs: 8 WMMAs per K-chunk per wave)
// K%32==0, M%64==0, N covered by grid. EXEC all-ones at every WMMA.
// Staging: global_load_async_to_lds_b128, ping-pong LDS, 1 barrier/chunk.
// ---------------------------------------------------------------------------
template<int NI>
__global__ __launch_bounds__(256)
void k_gemm_wmma(const __bf16* __restrict__ A, int lda,
                 const __bf16* __restrict__ Bt, int ldb,
                 const float*  __restrict__ bias,
                 const float*  __restrict__ addend,
                 float*        __restrict__ outF,
                 float*        __restrict__ vstate,
                 __bf16*       __restrict__ outS,
                 int ldc, int K, int mode)
{
  constexpr int MI = 2;                 // 16-tiles per wave along M
  constexpr int BM = 64;                // 2 waveM * MI * 16
  constexpr int BN = 64 * NI;           // 4 waveN * NI * 16
  // row stride 40 elems = 80B: every 16B async store stays 16B-aligned
  __shared__ __bf16 Alds[2][BM * 40] __attribute__((aligned(16)));
  __shared__ __bf16 Blds[2][BN * 40] __attribute__((aligned(16)));

  const int tid   = threadIdx.x;
  const int l     = tid & 31;
  const int wid   = tid >> 5;
  const int waveM = wid >> 2;              // 0..1
  const int waveN = wid & 3;               // 0..3
  const int rowBase = blockIdx.y * BM;
  const int colBase = blockIdx.x * BN;

  // staging: 16B (8 bf16) per async op; A: 1 seg/thread, B: NI segs/thread
  const int sr = tid >> 2;                 // 0..63
  const int cs = (tid & 3) << 3;           // 0,8,16,24

  const unsigned aSeg = (unsigned)(sr * 40 + cs) * 2u;
  unsigned bSeg[NI];
  const __bf16* bRow[NI];
#pragma unroll
  for (int i = 0; i < NI; ++i) {
    bSeg[i] = (unsigned)((sr + 64 * i) * 40 + cs) * 2u;
    bRow[i] = Bt + (long)(colBase + sr + 64 * i) * ldb + cs;
  }
  const unsigned aB[2] = { lds_off(&Alds[0][0]), lds_off(&Alds[1][0]) };
  const unsigned bB[2] = { lds_off(&Blds[0][0]), lds_off(&Blds[1][0]) };
  const __bf16* aRow = A + (long)(rowBase + sr) * lda + cs;

  auto issue = [&](int ch, int b) {
    const int k0 = ch << 5;
    async_load_b128(aB[b] + aSeg, aRow + k0);
#pragma unroll
    for (int i = 0; i < NI; ++i)
      async_load_b128(bB[b] + bSeg[i], bRow[i] + k0);
  };

  const int nch = K >> 5;
  issue(0, 0);

  v8f acc[MI][NI];
#pragma unroll
  for (int mi = 0; mi < MI; ++mi)
#pragma unroll
    for (int ni = 0; ni < NI; ++ni) acc[mi][ni] = (v8f){};

  const int m  = l & 15;
  const int kh = l >> 4;
  int mL[MI], nL[NI];
#pragma unroll
  for (int mi = 0; mi < MI; ++mi) mL[mi] = (waveM * MI * 16 + mi * 16 + m) * 40;
#pragma unroll
  for (int ni = 0; ni < NI; ++ni) nL[ni] = (waveN * NI * 16 + ni * 16 + m) * 40;

  for (int ch = 0; ch < nch; ++ch) {
    wait_async0();        // this wave's fills for buffer ch&1 have landed
    __syncthreads();      // everyone's fills have landed
    if (ch + 1 < nch) issue(ch + 1, (ch + 1) & 1);
    if (ch + 2 < nch) {   // warm L2 two chunks ahead (global_prefetch_b8)
      __builtin_prefetch(aRow + ((ch + 2) << 5), 0, 1);
      __builtin_prefetch(bRow[0] + ((ch + 2) << 5), 0, 1);
    }
    const int b = ch & 1;
    ABFrag af[MI], bf[NI];
#pragma unroll
    for (int v = 0; v < 8; ++v) {
      const int kk = ((v >> 2) << 4) | (kh << 3) | ((v & 3) << 1);
#pragma unroll
      for (int mi = 0; mi < MI; ++mi)
        af[mi].u[v] = *(const unsigned int*)&Alds[b][mL[mi] + kk];
#pragma unroll
      for (int ni = 0; ni < NI; ++ni)
        bf[ni].u[v] = *(const unsigned int*)&Blds[b][nL[ni] + kk];
    }
#pragma unroll
    for (int mi = 0; mi < MI; ++mi)
#pragma unroll
      for (int ni = 0; ni < NI; ++ni)
        acc[mi][ni] = __builtin_amdgcn_wmma_f32_16x16x32_bf16(
            false, af[mi].v, false, bf[ni].v, (short)0, acc[mi][ni], false, false);
  }

  // epilogue: lane holds C rows r0..r0+7 at one column per 16x16 tile
  const int r0 = (l >> 4) << 3;
#pragma unroll
  for (int mi = 0; mi < MI; ++mi) {
#pragma unroll
    for (int ni = 0; ni < NI; ++ni) {
      const int col = colBase + waveN * NI * 16 + ni * 16 + (l & 15);
      const float bcol = bias[col];
#pragma unroll
      for (int r = 0; r < 8; ++r) {
        const int  row = rowBase + waveM * MI * 16 + mi * 16 + r0 + r;
        const long idx = (long)row * ldc + col;
        float pre = acc[mi][ni][r] + bcol;
        if (mode == 0) {
          outF[idx] = pre;
        } else {
          if (addend) pre += addend[idx];
          float vm = vstate[idx];
          float h  = vm + (pre - vm) * 0.5f;           // TAU = 2
          float s  = (h - 1.0f >= 0.0f) ? 1.0f : 0.0f; // VTH = 1
          vstate[idx] = h * (1.0f - s);                // hard reset
          outS[idx]   = (__bf16)s;
        }
      }
    }
  }
}

// ---------------------------------------------------------------------------
// LayerNorm over H=1024 per row (block = 256 threads, 4 elems/thread)
// ---------------------------------------------------------------------------
__global__ __launch_bounds__(256)
void k_layernorm(float* __restrict__ y, const float* __restrict__ g,
                 const float* __restrict__ b)
{
  __shared__ float red[256];
  float* p = y + (long)blockIdx.x * Hh;
  const int tid = threadIdx.x;
  float vals[4];
  float s = 0.0f;
#pragma unroll
  for (int i = 0; i < 4; ++i) { float v = p[tid + i * 256]; vals[i] = v; s += v; }
  red[tid] = s; __syncthreads();
  for (int o = 128; o > 0; o >>= 1) { if (tid < o) red[tid] += red[tid + o]; __syncthreads(); }
  const float mu = red[0] * (1.0f / Hh);
  __syncthreads();
  float ss = 0.0f;
#pragma unroll
  for (int i = 0; i < 4; ++i) { float d = vals[i] - mu; ss += d * d; }
  red[tid] = ss; __syncthreads();
  for (int o = 128; o > 0; o >>= 1) { if (tid < o) red[tid] += red[tid + o]; __syncthreads(); }
  const float rstd = __frsqrt_rn(red[0] * (1.0f / Hh) + 1e-5f);
#pragma unroll
  for (int i = 0; i < 4; ++i) {
    const int c = tid + i * 256;
    p[c] = (vals[i] - mu) * rstd * g[c] + b[c];
  }
}

// ---------------------------------------------------------------------------
// Elementwise LIF scan over T (third LIF layer): thread = (n,h)
// ---------------------------------------------------------------------------
__global__ void k_lif_scan(const float* __restrict__ y, __bf16* __restrict__ sp) {
  const int i = blockIdx.x * 256 + threadIdx.x;
  if (i >= Nn * Hh) return;
  float v = 0.0f;
  for (int t = 0; t < Tn; ++t) {
    const float x = y[(long)t * Nn * Hh + i];
    const float h = v + (x - v) * 0.5f;
    const float s = (h - 1.0f >= 0.0f) ? 1.0f : 0.0f;
    v = h * (1.0f - s);
    sp[(long)t * Nn * Hh + i] = (__bf16)s;
  }
}

// ---------------------------------------------------------------------------
// Output: out[n,c] = b_out[c] + (1/T) * sum_t sum_h sp[t,n,h] * W_out[h,c]
// ---------------------------------------------------------------------------
__global__ void k_init_out(float* out, const float* __restrict__ b) {
  int i = blockIdx.x * 256 + threadIdx.x;
  if (i < Nn * NCLS) out[i] = b[i % NCLS];
}

__global__ __launch_bounds__(256)
void k_out_reduce(const __bf16* __restrict__ sp, const float* __restrict__ Wout,
                  float* __restrict__ out)
{
  __shared__ float sred[NCLS];
  const int tn  = blockIdx.x;          // t*Nn + n
  const int n   = tn % Nn;
  const int tid = threadIdx.x;
  const __bf16* row = sp + (long)tn * Hh;
  float acc[NCLS];
#pragma unroll
  for (int c = 0; c < NCLS; ++c) acc[c] = 0.0f;
  for (int h = tid; h < Hh; h += 256) {
    const float s = (float)row[h];
    if (s != 0.0f) {                    // spikes are 0/1
      const float* w = Wout + (long)h * NCLS;
#pragma unroll
      for (int c = 0; c < NCLS; ++c) acc[c] += w[c];
    }
  }
  if (tid < NCLS) sred[tid] = 0.0f;
  __syncthreads();
#pragma unroll
  for (int c = 0; c < NCLS; ++c) atomicAdd(&sred[c], acc[c]);
  __syncthreads();
  if (tid < NCLS) atomicAdd(&out[n * NCLS + tid], sred[tid] * (1.0f / Tn));
}

// ---------------------------------------------------------------------------
// Host orchestration
// ---------------------------------------------------------------------------
extern "C" void kernel_launch(void* const* d_in, const int* in_sizes, int n_in,
                              void* d_out, int out_size, void* d_ws, size_t ws_size,
                              hipStream_t stream)
{
  (void)in_sizes; (void)n_in; (void)out_size; (void)ws_size;
  const float* x     = (const float*)d_in[0];
  const float* W_in  = (const float*)d_in[1];
  const float* b_in  = (const float*)d_in[2];
  const float* W_rec = (const float*)d_in[3];
  const float* b_rec = (const float*)d_in[4];
  const float* W2    = (const float*)d_in[5];
  const float* b2    = (const float*)d_in[6];
  const float* W3    = (const float*)d_in[7];
  const float* b3    = (const float*)d_in[8];
  const float* ln_g  = (const float*)d_in[9];
  const float* ln_b  = (const float*)d_in[10];
  const float* W_out = (const float*)d_in[11];
  const float* b_out = (const float*)d_in[12];
  float* out = (float*)d_out;

  char* ws = (char*)d_ws;
  size_t off = 0;
  auto alloc = [&](size_t bytes) -> char* {
    char* p = ws + off;
    off += (bytes + 255) & ~(size_t)255;
    return p;
  };
  __bf16* xbf   = (__bf16*)alloc((size_t)Tn * Nn * Cp * 2);   // ~36.0 MB
  __bf16* WinT  = (__bf16*)alloc((size_t)Hh * Cp * 2);        // [H][Cp]
  __bf16* WrecT = (__bf16*)alloc((size_t)Hh * Hh * 2);        // [H][H] transposed
  __bf16* W2T   = (__bf16*)alloc((size_t)Hh * Hh * 2);
  __bf16* W3T   = (__bf16*)alloc((size_t)Hh * Hh * 2);
  float*  xin   = (float*)alloc((size_t)Tn * Nn * Hh * 4);    // ~104.9 MB; reused as y
  __bf16* sA    = (__bf16*)alloc((size_t)Nn * Hh * 2);
  __bf16* sB    = (__bf16*)alloc((size_t)Nn * Hh * 2);
  float*  v1    = (float*)alloc((size_t)Nn * Hh * 4);
  float*  v2    = (float*)alloc((size_t)Nn * Hh * 4);
  __bf16* feats = (__bf16*)alloc((size_t)Tn * Nn * Hh * 2);   // ~52.4 MB
  __bf16* sp3   = (__bf16*)alloc((size_t)Tn * Nn * Hh * 2);   // ~52.4 MB

  // --- one-time precision conversion + weight transposition ---
  k_conv_x<<<(Tn * Nn * Cp + 255) / 256, 256, 0, stream>>>(x, xbf);
  k_transpose_w<<<(Hh * Cp + 255) / 256, 256, 0, stream>>>(W_in, WinT, Cc, Hh, Cp);
  k_transpose_w<<<(Hh * Hh + 255) / 256, 256, 0, stream>>>(W_rec, WrecT, Hh, Hh, Hh);
  k_transpose_w<<<(Hh * Hh + 255) / 256, 256, 0, stream>>>(W2, W2T, Hh, Hh, Hh);
  k_transpose_w<<<(Hh * Hh + 255) / 256, 256, 0, stream>>>(W3, W3T, Hh, Hh, Hh);

  // --- state init ---
  k_zero_f32 <<<(Nn * Hh + 255) / 256, 256, 0, stream>>>(v1, Nn * Hh);
  k_zero_f32 <<<(Nn * Hh + 255) / 256, 256, 0, stream>>>(v2, Nn * Hh);
  k_zero_bf16<<<(Nn * Hh + 255) / 256, 256, 0, stream>>>(sA, Nn * Hh);

  // --- input projection: xin[T*N, H] = xbf @ W_in + b_in (batched over T) ---
  dim3 gBig(Hh / 256, (Tn * Nn) / 64);           // 4 x 400
  k_gemm_wmma<4><<<gBig, 256, 0, stream>>>(xbf, Cp, WinT, Cp, b_in,
                                           nullptr, xin, nullptr, nullptr,
                                           Hh, Cp, 0);

  // --- recurrent LIF scan over T (sequential, fused GEMM+LIF per step) ---
  dim3 gStep(Hh / 128, Nn / 64);                  // 8 x 4
  for (int t = 0; t < Tn; ++t) {
    __bf16* sprev = (t & 1) ? sB : sA;
    __bf16* snext = (t & 1) ? sA : sB;
    // inp = xin_t + s_prev @ W_rec + b_rec ; (v1,s1) = LIF(v1, inp)
    k_gemm_wmma<2><<<gStep, 256, 0, stream>>>(sprev, Hh, WrecT, Hh, b_rec,
                                              xin + (size_t)t * Nn * Hh,
                                              nullptr, v1, snext, Hh, Hh, 1);
    // (v2, feat_t) = LIF(v2, s1 @ W2 + b2)
    k_gemm_wmma<2><<<gStep, 256, 0, stream>>>(snext, Hh, W2T, Hh, b2,
                                              nullptr, nullptr, v2,
                                              feats + (size_t)t * Nn * Hh, Hh, Hh, 1);
  }

  // --- part2: y = feats @ W3 + b3 (reuse xin buffer), LayerNorm, LIF, out ---
  float* y = xin;
  k_gemm_wmma<4><<<gBig, 256, 0, stream>>>(feats, Hh, W3T, Hh, b3,
                                           nullptr, y, nullptr, nullptr,
                                           Hh, Hh, 0);
  k_layernorm<<<Tn * Nn, 256, 0, stream>>>(y, ln_g, ln_b);
  k_lif_scan<<<(Nn * Hh + 255) / 256, 256, 0, stream>>>(y, sp3);
  k_init_out<<<(Nn * NCLS + 255) / 256, 256, 0, stream>>>(out, b_out);
  k_out_reduce<<<Tn * Nn, 256, 0, stream>>>(sp3, W_out, out);
}
